// GCoRe_78262894068282
// MI455X (gfx1250) — matrix operand order repaired and checked by reading, not compiled, not run on hardware
//
#include <hip/hip_runtime.h>
#include <hip/hip_bf16.h>

#define DIM      128
#define NLAYERS  8
#define NCLASSES 10
#define NGRAPHS  64

typedef __attribute__((ext_vector_type(16))) __bf16 v16bf;
typedef __attribute__((ext_vector_type(8)))  __bf16 v8bf;
typedef __attribute__((ext_vector_type(8)))  float  v8f;

#define WS_STRIDE 136   // row start 272B (16B aligned), k-offsets 32B aligned,
                        // 68-dword row stride -> conflict-free b128 reads

// ---------------------------------------------------------------------------
// Edge scatter-add: z[dst] += h[src].  One wave (32 lanes) per edge, 4 floats
// per lane -> float4 gather + 4x global_atomic_add_f32 (L2 resident).
// ---------------------------------------------------------------------------
__global__ void scatter_add_kernel(const float* __restrict__ h,
                                   const int* __restrict__ esrc,
                                   const int* __restrict__ edst,
                                   float* __restrict__ z, int nEdges) {
    int idx  = blockIdx.x * blockDim.x + threadIdx.x;
    int e    = idx >> 5;
    if (e >= nEdges) return;
    int lane = idx & 31;
    int s = esrc[e], d = edst[e];
    float4 v = *reinterpret_cast<const float4*>(h + (size_t)s * DIM + lane * 4);
    float* zp = z + (size_t)d * DIM + lane * 4;
    atomicAdd(zp + 0, v.x);
    atomicAdd(zp + 1, v.y);
    atomicAdd(zp + 2, v.z);
    atomicAdd(zp + 3, v.w);
}

// ---------------------------------------------------------------------------
// [M,128] x [128,128] GEMM via v_wmma_f32_16x16x32_bf16 (f32 accumulate).
// Block = 256 threads = 8 waves; block owns 128 rows; each wave 16 rows x 128
// cols (8 C-tiles of 16x16).  W transposed into LDS as bf16 (native cvt); per
// the contiguous k order used here (k = 16*half + e) each lane's fragment is 2 contiguous 16B LDS spans
// -> 2x ds_load_b128.  A fragment = 2 contiguous 32B global spans, converted
// with native (packed) f32->bf16 casts.
// RELU: fused bias+leaky-relu (MLP layer 1).
// STATS: fused bias + per-column sum/sumsq atomics for BatchNorm (layer 2).
// In-place (Out == A) is safe: block touches only its own rows, reads first.
// ---------------------------------------------------------------------------
template<bool RELU, bool STATS>
__global__ __launch_bounds__(256) void gemm128_kernel(
    const float* __restrict__ A, const float* __restrict__ W,
    const float* __restrict__ bias, float* __restrict__ Out,
    float* __restrict__ colSum, float* __restrict__ colSumSq, int M)
{
    __shared__ __attribute__((aligned(32))) __bf16 Ws[DIM][WS_STRIDE]; // Ws[n][k]

    const int tid = threadIdx.x;
    {   // cooperative transposed load of W (row-major [k][n]) into LDS bf16
        int k  = tid >> 1;
        int n0 = (tid & 1) * 64;
        const float4* wp = reinterpret_cast<const float4*>(W + k * DIM + n0);
        #pragma unroll
        for (int j = 0; j < 16; ++j) {
            float4 w = wp[j];
            int n = n0 + j * 4;
            Ws[n + 0][k] = (__bf16)w.x;
            Ws[n + 1][k] = (__bf16)w.y;
            Ws[n + 2][k] = (__bf16)w.z;
            Ws[n + 3][k] = (__bf16)w.w;
        }
    }
    __syncthreads();

    const int lane    = tid & 31;
    const int wave    = tid >> 5;
    const int rowBase = blockIdx.x * 128 + wave * 16;
    const int lrow    = lane & 15;
    const int lhalf   = lane >> 4;

    int arow  = rowBase + lrow;
    int arowC = arow < M ? arow : (M - 1);          // clamp; masked on store
    const float* Arow = A + (size_t)arowC * DIM;

    v8f acc[8];
    #pragma unroll
    for (int nt = 0; nt < 8; ++nt)
        #pragma unroll
        for (int j = 0; j < 8; ++j) acc[nt][j] = 0.0f;

    const int kgrp = lhalf * 16;     // A-layout K sub-offset for this lane half
    #pragma unroll
    for (int kt = 0; kt < 4; ++kt) {
        const int kb = kt * 32;
        // A fragment 16x32 bf16: element e -> K = kb + kgrp + e (e<16).
        float tmp[16];
        {
            const float4* a0 = reinterpret_cast<const float4*>(Arow + kb + kgrp);
            const float4* a1 = reinterpret_cast<const float4*>(Arow + kb + kgrp + 8);
            *reinterpret_cast<float4*>(&tmp[0])  = a0[0];
            *reinterpret_cast<float4*>(&tmp[4])  = a0[1];
            *reinterpret_cast<float4*>(&tmp[8])  = a1[0];
            *reinterpret_cast<float4*>(&tmp[12]) = a1[1];
        }
        v16bf afrag;
        #pragma unroll
        for (int e = 0; e < 16; ++e) afrag[e] = (__bf16)tmp[e];

        const int bk0 = kb + lhalf * 16;   // B-layout: lanes 16-31 hold K+16..K+31
        #pragma unroll
        for (int nt = 0; nt < 8; ++nt) {
            const int bn = nt * 16 + lrow; // B column = lane&15
            // lane's 16 B values are contiguous in Ws row -> two b128 loads
            v8bf lo = *reinterpret_cast<const v8bf*>(&Ws[bn][bk0]);
            v8bf hi = *reinterpret_cast<const v8bf*>(&Ws[bn][bk0 + 8]);
            v16bf bfrag = __builtin_shufflevector(lo, hi,
                0, 1, 2, 3, 4, 5, 6, 7, 8, 9, 10, 11, 12, 13, 14, 15);
            acc[nt] = __builtin_amdgcn_wmma_f32_16x16x32_bf16(
                false, afrag, false, bfrag, (short)0, acc[nt], false, false);
        }
    }

    // epilogue: C layout -> lane l, VGPR r: M = r + 8*(l>>4), N = l&15
    #pragma unroll
    for (int nt = 0; nt < 8; ++nt) {
        const int n  = nt * 16 + lrow;
        const float bv = bias[n];
        float s = 0.0f, s2 = 0.0f;
        #pragma unroll
        for (int r = 0; r < 8; ++r) {
            int m = rowBase + r + 8 * lhalf;
            if (m < M) {
                float v = acc[nt][r] + bv;
                if (RELU) v = v > 0.0f ? v : 0.01f * v;
                Out[(size_t)m * DIM + n] = v;
                if (STATS) { s += v; s2 += v * v; }
            }
        }
        if (STATS) {
            atomicAdd(&colSum[n], s);
            atomicAdd(&colSumSq[n], s2);
        }
    }
}

// ---------------------------------------------------------------------------
__global__ void zero_kernel(float* __restrict__ p, int n) {
    int i = blockIdx.x * blockDim.x + threadIdx.x;
    if (i < n) p[i] = 0.0f;
}

__global__ void bn_finalize_kernel(const float* __restrict__ colSum,
                                   const float* __restrict__ colSumSq,
                                   const float* __restrict__ gamma,
                                   const float* __restrict__ beta,
                                   float* __restrict__ scale,
                                   float* __restrict__ shift, float invN) {
    int n = threadIdx.x;
    if (n >= DIM) return;
    float mu  = colSum[n] * invN;
    float var = colSumSq[n] * invN - mu * mu;
    float sc  = gamma[n] * rsqrtf(var + 1e-5f);
    scale[n] = sc;
    shift[n] = beta[n] - mu * sc;
}

__global__ void bn_apply_kernel(const float* __restrict__ z,
                                const float* __restrict__ scale,
                                const float* __restrict__ shift,
                                float* __restrict__ out, int total4, int relu) {
    int i = blockIdx.x * blockDim.x + threadIdx.x;
    if (i >= total4) return;
    int n0 = (i * 4) & (DIM - 1);
    float4 v = reinterpret_cast<const float4*>(z)[i];
    float4 r;
    r.x = v.x * scale[n0 + 0] + shift[n0 + 0];
    r.y = v.y * scale[n0 + 1] + shift[n0 + 1];
    r.z = v.z * scale[n0 + 2] + shift[n0 + 2];
    r.w = v.w * scale[n0 + 3] + shift[n0 + 3];
    if (relu) {
        r.x = r.x > 0.0f ? r.x : 0.01f * r.x;
        r.y = r.y > 0.0f ? r.y : 0.01f * r.y;
        r.z = r.z > 0.0f ? r.z : 0.01f * r.z;
        r.w = r.w > 0.0f ? r.w : 0.01f * r.w;
    }
    reinterpret_cast<float4*>(out)[i] = r;
}

// ---------------------------------------------------------------------------
// Per-node softmax(h + gumbel) over 128 dims (wave per node, wave32 shuffles)
// + global_add_pool into pooled[64][128] via atomics.
// ---------------------------------------------------------------------------
__global__ void softmax_pool_kernel(const float* __restrict__ h,
                                    const float* __restrict__ gum,
                                    const int* __restrict__ batch,
                                    float* __restrict__ cOut,
                                    float* __restrict__ pooled, int N) {
    int node = blockIdx.x * 8 + (threadIdx.x >> 5);
    if (node >= N) return;
    int lane = threadIdx.x & 31;
    size_t base = (size_t)node * DIM + lane * 4;
    float4 hv = *reinterpret_cast<const float4*>(h + base);
    float4 gv = *reinterpret_cast<const float4*>(gum + base);
    float t0 = hv.x + gv.x, t1 = hv.y + gv.y, t2 = hv.z + gv.z, t3 = hv.w + gv.w;
    float mx = fmaxf(fmaxf(t0, t1), fmaxf(t2, t3));
    #pragma unroll
    for (int o = 16; o > 0; o >>= 1) mx = fmaxf(mx, __shfl_xor(mx, o, 32));
    float e0 = __expf(t0 - mx), e1 = __expf(t1 - mx);
    float e2 = __expf(t2 - mx), e3 = __expf(t3 - mx);
    float sum = e0 + e1 + e2 + e3;
    #pragma unroll
    for (int o = 16; o > 0; o >>= 1) sum += __shfl_xor(sum, o, 32);
    float inv = 1.0f / sum;
    float4 c;
    c.x = e0 * inv; c.y = e1 * inv; c.z = e2 * inv; c.w = e3 * inv;
    *reinterpret_cast<float4*>(cOut + base) = c;
    int b = batch[node];
    float* pp = pooled + b * DIM + lane * 4;
    atomicAdd(pp + 0, c.x);
    atomicAdd(pp + 1, c.y);
    atomicAdd(pp + 2, c.z);
    atomicAdd(pp + 3, c.w);
}

// ---------------------------------------------------------------------------
// Tiny dense head: t1 = lrelu(pooled @ Wd1 + bd1); out = softmax(t1 @ Wd2+bd2)
// ---------------------------------------------------------------------------
__global__ void head1_kernel(const float* __restrict__ pooled,
                             const float* __restrict__ Wd1,
                             const float* __restrict__ bd1,
                             float* __restrict__ t1) {
    int i = blockIdx.x * blockDim.x + threadIdx.x;
    if (i >= NGRAPHS * DIM) return;
    int g = i >> 7, n = i & (DIM - 1);
    float acc = bd1[n];
    for (int k = 0; k < DIM; ++k) acc += pooled[g * DIM + k] * Wd1[k * DIM + n];
    t1[i] = acc > 0.0f ? acc : 0.01f * acc;
}

__global__ void head2_kernel(const float* __restrict__ t1,
                             const float* __restrict__ Wd2,
                             const float* __restrict__ bd2,
                             float* __restrict__ out) {
    int g = threadIdx.x;
    if (g >= NGRAPHS) return;
    float logits[NCLASSES];
    float mx = -1e30f;
    for (int c = 0; c < NCLASSES; ++c) {
        float acc = bd2[c];
        for (int k = 0; k < DIM; ++k) acc += t1[g * DIM + k] * Wd2[k * NCLASSES + c];
        logits[c] = acc;
        mx = fmaxf(mx, acc);
    }
    float sum = 0.0f;
    for (int c = 0; c < NCLASSES; ++c) { logits[c] = __expf(logits[c] - mx); sum += logits[c]; }
    float inv = 1.0f / sum;
    for (int c = 0; c < NCLASSES; ++c) out[g * NCLASSES + c] = logits[c] * inv;
}

// ---------------------------------------------------------------------------
extern "C" void kernel_launch(void* const* d_in, const int* in_sizes, int n_in,
                              void* d_out, int out_size, void* d_ws, size_t ws_size,
                              hipStream_t stream) {
    const float* x      = (const float*)d_in[0];
    const int*   eidx   = (const int*)  d_in[1];
    const int*   batch  = (const int*)  d_in[2];
    const float* gum    = (const float*)d_in[3];
    const float* W1s    = (const float*)d_in[4];
    const float* b1s    = (const float*)d_in[5];
    const float* W2s    = (const float*)d_in[6];
    const float* b2s    = (const float*)d_in[7];
    const float* gammas = (const float*)d_in[8];
    const float* betas  = (const float*)d_in[9];
    const float* Wd1    = (const float*)d_in[10];
    const float* bd1    = (const float*)d_in[11];
    const float* Wd2    = (const float*)d_in[12];
    const float* bd2    = (const float*)d_in[13];

    const int N      = in_sizes[0] / DIM;
    const int nEdges = in_sizes[1] / 2;
    const int* esrc = eidx;
    const int* edst = eidx + nEdges;

    const size_t featBytes = (size_t)N * DIM * sizeof(float);  // 25.6 MB
    char* ws = (char*)d_ws;
    float* bufA     = (float*)(ws);                       // h (next)
    float* bufB     = (float*)(ws + featBytes);           // z / mid / z2 (in-place)
    float* colSum   = (float*)(ws + 2 * featBytes);       // 128
    float* colSumSq = colSum + DIM;                       // 128 (contiguous with colSum)
    float* scale    = colSumSq + DIM;
    float* shift    = scale + DIM;
    float* pooled   = shift + DIM;                        // 64*128
    float* t1       = pooled + NGRAPHS * DIM;             // 64*128

    float* outHead  = (float*)d_out;                      // [64,10]
    float* outC     = outHead + NGRAPHS * NCLASSES;       // [50000,128]

    const int gemmBlocks    = (N + 127) / 128;
    const int scatterBlocks = (nEdges * 32 + 255) / 256;
    const int elem4Blocks   = (N * (DIM / 4) + 255) / 256;
    const int smaxBlocks    = (N + 7) / 8;

    for (int layer = 0; layer < NLAYERS; ++layer) {
        const float* h = (layer == 0) ? x : bufA;
        // z = h  (GIN eps=0), then z += sum of neighbors
        hipMemcpyAsync(bufB, h, featBytes, hipMemcpyDeviceToDevice, stream);
        scatter_add_kernel<<<scatterBlocks, 256, 0, stream>>>(h, esrc, edst, bufB, nEdges);
        // mid = lrelu(z @ W1 + b1)   (in-place in bufB)
        gemm128_kernel<true, false><<<gemmBlocks, 256, 0, stream>>>(
            bufB, W1s + (size_t)layer * DIM * DIM, b1s + layer * DIM,
            bufB, nullptr, nullptr, N);
        // z2 = mid @ W2 + b2, with fused BN-stat accumulation (in-place)
        zero_kernel<<<1, 256, 0, stream>>>(colSum, 2 * DIM);
        gemm128_kernel<false, true><<<gemmBlocks, 256, 0, stream>>>(
            bufB, W2s + (size_t)layer * DIM * DIM, b2s + layer * DIM,
            bufB, colSum, colSumSq, N);
        // batchnorm + (lrelu except last layer)
        bn_finalize_kernel<<<1, DIM, 0, stream>>>(
            colSum, colSumSq, gammas + layer * DIM, betas + layer * DIM,
            scale, shift, 1.0f / (float)N);
        bn_apply_kernel<<<elem4Blocks, 256, 0, stream>>>(
            bufB, scale, shift, bufA, N * (DIM / 4), layer < NLAYERS - 1 ? 1 : 0);
    }

    // gumbel softmax per node + global add pool
    zero_kernel<<<(NGRAPHS * DIM + 255) / 256, 256, 0, stream>>>(pooled, NGRAPHS * DIM);
    softmax_pool_kernel<<<smaxBlocks, 256, 0, stream>>>(bufA, gum, batch, outC, pooled, N);

    // dense head
    head1_kernel<<<(NGRAPHS * DIM + 255) / 256, 256, 0, stream>>>(pooled, Wd1, bd1, t1);
    head2_kernel<<<1, NGRAPHS, 0, stream>>>(t1, Wd2, bd2, outHead);
}
